// Nonlinear_59167469470253
// MI455X (gfx1250) — matrix operand order repaired and checked by reading, not compiled, not run on hardware
//
#include <hip/hip_runtime.h>

// ---------------------------------------------------------------------------
// MI455X (gfx1250) fused GLU-MLP + group-RMS norm.
// Fast path (if ws fits): one-time bf16 convert of z + bf16 [N][K] transpose of
// W1/W2/W3 (~35us of bandwidth), then GEMM hot loops are pure
// global_load_async_to_lds_b128 DMA + ds_load_b128 fragments + v_wmma bf16.
// Fallback path: convert-in-loop kernels (round-2 structure).
//   K1: gated = sigmoid(z@W1+b1) * (z@W2+b2)   (dual bf16-WMMA GEMM, bf16 out)
//   K2: out   = rivalry_norm(gated@W3 + b3)    (bf16-WMMA GEMM, fused norm)
// Block tile 128x128x32, 256 threads = 8 waves (wave32), wave tile 64x32.
// ---------------------------------------------------------------------------

typedef __attribute__((ext_vector_type(16))) __bf16 v16bf;
typedef __attribute__((ext_vector_type(8)))  float  v8f;

#define BM 128
#define BN 128
#define BK 32
#define LP 40    // LDS row pitch in bf16 elems (80B = 5*16B -> conflict-free b128)
#define LPB 80   // pitch in bytes

union FragB16 { uint4 u4[2]; v16bf v; };

// ---- f32 -> bf16 (RNE): 2x v_add3 + 1x v_perm_b32 merge of the high halves
__device__ __forceinline__ unsigned pack2bf(float lo, float hi) {
    unsigned a = __float_as_uint(lo), b = __float_as_uint(hi);
    a += 0x7FFFu + ((a >> 16) & 1u);
    b += 0x7FFFu + ((b >> 16) & 1u);
    // D = {b[31:16], a[31:16]} : sel bytes {7,6,3,2} of {S0=b, S1=a}
    return __builtin_amdgcn_perm(b, a, 0x07060302u);
}
__device__ __forceinline__ unsigned short f2bf(float x) {
    unsigned u = __float_as_uint(x);
    u += 0x7FFFu + ((u >> 16) & 1u);
    return (unsigned short)(u >> 16);
}

// ---- CDNA5 async global->LDS (no VGPR round trip, ASYNCcnt tracked) ----
__device__ __forceinline__ void async_copy_b128(void* lds_ptr, const void* gptr) {
    unsigned l = (unsigned)(unsigned long long)(uintptr_t)lds_ptr; // low 32b of flat = LDS offset
    unsigned long long g = (unsigned long long)(uintptr_t)gptr;
    asm volatile("global_load_async_to_lds_b128 %0, %1, off"
                 :: "v"(l), "v"(g) : "memory");
}
__device__ __forceinline__ void wait_async0() {
#if defined(__has_builtin) && __has_builtin(__builtin_amdgcn_s_wait_asynccnt)
    __builtin_amdgcn_s_wait_asynccnt(0);
#else
    asm volatile("s_wait_asynccnt 0" ::: "memory");
#endif
}

// ===========================================================================
// Preprocessing kernels (fast path)
// ===========================================================================

// streaming f32 -> bf16 (z), 1 float4 per thread
__global__ __launch_bounds__(256)
void convert_bf16(const float* __restrict__ src, unsigned short* __restrict__ dst)
{
    size_t i = ((size_t)blockIdx.x * 256 + threadIdx.x) * 4;
    float4 v = *reinterpret_cast<const float4*>(src + i);
    uint2 p; p.x = pack2bf(v.x, v.y); p.y = pack2bf(v.z, v.w);
    *reinterpret_cast<uint2*>(dst + i) = p;
}

// f32 [K][N] -> bf16 [N][K] via LDS (64x64 tile per block)
__global__ __launch_bounds__(256)
void transpose_to_bf16(const float* __restrict__ src, unsigned short* __restrict__ dst,
                       int K, int N)
{
    __shared__ __attribute__((aligned(16))) unsigned short t[64 * 80]; // pitch 160B
    const int tid = threadIdx.x;
    const int k0 = blockIdx.y * 64, n0 = blockIdx.x * 64;
#pragma unroll
    for (int r = 0; r < 4; ++r) {
        int idx = tid + r * 256;          // 64x64 f32 = 1024 float4
        int k = idx >> 4, c4 = idx & 15;
        float4 v = *reinterpret_cast<const float4*>(src + (size_t)(k0 + k) * N + n0 + c4 * 4);
        int nb = c4 * 4;
        t[(nb + 0) * 80 + k] = f2bf(v.x);
        t[(nb + 1) * 80 + k] = f2bf(v.y);
        t[(nb + 2) * 80 + k] = f2bf(v.z);
        t[(nb + 3) * 80 + k] = f2bf(v.w);
    }
    __syncthreads();
#pragma unroll
    for (int r = 0; r < 2; ++r) {
        int idx = tid + r * 256;          // 64x64 bf16 = 512 uint4
        int n = idx >> 2, k16 = idx & 3;
        uint4 v = *reinterpret_cast<const uint4*>(
            reinterpret_cast<const char*>(t) + n * 160 + k16 * 16);
        *reinterpret_cast<uint4*>(dst + (size_t)(n0 + n) * K + k0 + k16 * 8) = v;
    }
}

// ===========================================================================
// Fast-path GEMM kernels: all tiles DMA'd with global_load_async_to_lds_b128.
// A source: bf16 row-major [rows][K]; B source: bf16 transposed [N][K].
// Both land in LDS as [row][k] with pitch LP (identical fragment addressing).
// ===========================================================================

__device__ __forceinline__ void async_tile(unsigned short* lds,
                                           const unsigned short* g,
                                           int base, size_t ldK, int kb, int tid)
{
#pragma unroll
    for (int r = 0; r < 2; ++r) {
        int idx = tid + r * 256;          // 128x32 bf16 = 512 b128 chunks
        int row = idx >> 2, k8 = idx & 3;
        async_copy_b128(reinterpret_cast<char*>(lds) + row * LPB + k8 * 16,
                        g + (size_t)(base + row) * ldK + kb + k8 * 8);
    }
}

__global__ __launch_bounds__(256)
void dual_gemm_gate_fast(const unsigned short* __restrict__ zb,
                         const unsigned short* __restrict__ w1t,
                         const unsigned short* __restrict__ w2t,
                         const float* __restrict__ b1, const float* __restrict__ b2,
                         unsigned short* __restrict__ gated,
                         int M, int K, int N)
{
    __shared__ __attribute__((aligned(16))) unsigned short sA [2][BM * LP];
    __shared__ __attribute__((aligned(16))) unsigned short sB1[2][BN * LP];
    __shared__ __attribute__((aligned(16))) unsigned short sB2[2][BN * LP];

    const int tid  = threadIdx.x;
    const int lane = tid & 31, wave = tid >> 5;
    const int wm = wave >> 2, wn = wave & 3;
    const int lr = lane & 15, lh = lane >> 4;
    const int m0 = blockIdx.y * BM, n0 = blockIdx.x * BN;

    v8f acc1[4][2], acc2[4][2];
#pragma unroll
    for (int i = 0; i < 4; ++i)
#pragma unroll
        for (int j = 0; j < 2; ++j)
#pragma unroll
            for (int e = 0; e < 8; ++e) { acc1[i][j][e] = 0.f; acc2[i][j][e] = 0.f; }

    auto prefetch = [&](int kb, int buf) {
        async_tile(sA [buf], zb,  m0, (size_t)K, kb, tid);
        async_tile(sB1[buf], w1t, n0, (size_t)K, kb, tid);
        async_tile(sB2[buf], w2t, n0, (size_t)K, kb, tid);
    };
    auto compute = [&](int buf) {
        FragB16 aF[4], b1F[2], b2F[2];
#pragma unroll
        for (int i = 0; i < 4; ++i) {
            int row = wm * 64 + i * 16 + lr;
            const char* base = reinterpret_cast<const char*>(sA[buf]) + row * LPB + lh * 32;
            aF[i].u4[0] = *reinterpret_cast<const uint4*>(base);       // K = 16h..16h+7
            aF[i].u4[1] = *reinterpret_cast<const uint4*>(base + 16);  // K = 16h+8..
        }
#pragma unroll
        for (int jn = 0; jn < 2; ++jn) {
            int col = wn * 32 + jn * 16 + lr;
            const char* c1 = reinterpret_cast<const char*>(sB1[buf]) + col * LPB + lh * 32;
            b1F[jn].u4[0] = *reinterpret_cast<const uint4*>(c1);       // K = 16h..16h+15
            b1F[jn].u4[1] = *reinterpret_cast<const uint4*>(c1 + 16);
            const char* c2 = reinterpret_cast<const char*>(sB2[buf]) + col * LPB + lh * 32;
            b2F[jn].u4[0] = *reinterpret_cast<const uint4*>(c2);
            b2F[jn].u4[1] = *reinterpret_cast<const uint4*>(c2 + 16);
        }
#pragma unroll
        for (int i = 0; i < 4; ++i)
#pragma unroll
            for (int jn = 0; jn < 2; ++jn) {
                acc1[i][jn] = __builtin_amdgcn_wmma_f32_16x16x32_bf16(
                    false, aF[i].v, false, b1F[jn].v, (short)0, acc1[i][jn], false, false);
                acc2[i][jn] = __builtin_amdgcn_wmma_f32_16x16x32_bf16(
                    false, aF[i].v, false, b2F[jn].v, (short)0, acc2[i][jn], false, false);
            }
    };

    const int nIter = K / BK;
    prefetch(0, 0);
    wait_async0();
    __syncthreads();
    for (int it = 0; it < nIter; ++it) {
        const int cur = it & 1;
        if (it + 1 < nIter) prefetch((it + 1) * BK, cur ^ 1);  // DMA overlaps WMMA
        compute(cur);
        wait_async0();
        __syncthreads();
    }

    // ---- epilogue: bias, GLU gate, bf16 store ----
#pragma unroll
    for (int jn = 0; jn < 2; ++jn) {
        int col = n0 + wn * 32 + jn * 16 + lr;
        float bb1 = b1[col], bb2 = b2[col];
#pragma unroll
        for (int i = 0; i < 4; ++i) {
            int rbase = m0 + wm * 64 + i * 16 + lh * 8;
#pragma unroll
            for (int j = 0; j < 8; ++j) {
                float g = acc1[i][jn][j] + bb1;
                float c = acc2[i][jn][j] + bb2;
                float val = c * (1.0f / (1.0f + __expf(-g)));
                gated[(size_t)(rbase + j) * N + col] = f2bf(val);
            }
        }
    }
}

__global__ __launch_bounds__(256)
void gemm_norm_fast(const unsigned short* __restrict__ gated,
                    const unsigned short* __restrict__ w3t,
                    const float* __restrict__ b3, const float* __restrict__ gamma,
                    float* __restrict__ out,
                    int M, int K, int N)
{
    __shared__ __attribute__((aligned(16))) unsigned short sA[2][BM * LP];
    __shared__ __attribute__((aligned(16))) unsigned short sB[2][BN * LP];
    __shared__ float sSum[BM * 4];
    __shared__ float sInv[BM];

    const int tid  = threadIdx.x;
    const int lane = tid & 31, wave = tid >> 5;
    const int wm = wave >> 2, wn = wave & 3;
    const int lr = lane & 15, lh = lane >> 4;
    const int m0 = blockIdx.y * BM, n0 = blockIdx.x * BN;

    v8f acc[4][2];
#pragma unroll
    for (int i = 0; i < 4; ++i)
#pragma unroll
        for (int j = 0; j < 2; ++j)
#pragma unroll
            for (int e = 0; e < 8; ++e) acc[i][j][e] = 0.f;

    auto prefetch = [&](int kb, int buf) {
        async_tile(sA[buf], gated, m0, (size_t)K, kb, tid);
        async_tile(sB[buf], w3t,   n0, (size_t)K, kb, tid);
    };
    auto compute = [&](int buf) {
        FragB16 aF[4], bF[2];
#pragma unroll
        for (int i = 0; i < 4; ++i) {
            int row = wm * 64 + i * 16 + lr;
            const char* base = reinterpret_cast<const char*>(sA[buf]) + row * LPB + lh * 32;
            aF[i].u4[0] = *reinterpret_cast<const uint4*>(base);
            aF[i].u4[1] = *reinterpret_cast<const uint4*>(base + 16);
        }
#pragma unroll
        for (int jn = 0; jn < 2; ++jn) {
            int col = wn * 32 + jn * 16 + lr;
            const char* cb = reinterpret_cast<const char*>(sB[buf]) + col * LPB + lh * 32;
            bF[jn].u4[0] = *reinterpret_cast<const uint4*>(cb);
            bF[jn].u4[1] = *reinterpret_cast<const uint4*>(cb + 16);
        }
#pragma unroll
        for (int i = 0; i < 4; ++i)
#pragma unroll
            for (int jn = 0; jn < 2; ++jn)
                acc[i][jn] = __builtin_amdgcn_wmma_f32_16x16x32_bf16(
                    false, aF[i].v, false, bF[jn].v, (short)0, acc[i][jn], false, false);
    };

    const int nIter = K / BK;
    prefetch(0, 0);
    wait_async0();
    __syncthreads();
    for (int it = 0; it < nIter; ++it) {
        const int cur = it & 1;
        if (it + 1 < nIter) prefetch((it + 1) * BK, cur ^ 1);
        compute(cur);
        wait_async0();
        __syncthreads();
    }

    // ---- epilogue: bias add ----
#pragma unroll
    for (int jn = 0; jn < 2; ++jn) {
        int col = n0 + wn * 32 + jn * 16 + lr;
        float bb = b3[col];
#pragma unroll
        for (int i = 0; i < 4; ++i)
#pragma unroll
            for (int j = 0; j < 8; ++j) acc[i][jn][j] += bb;
    }

    // ---- cross-lane sum of squares per row (group = the 128 cols of tile) ----
#pragma unroll
    for (int i = 0; i < 4; ++i) {
#pragma unroll
        for (int j = 0; j < 8; ++j) {
            float s = acc[i][0][j] * acc[i][0][j] + acc[i][1][j] * acc[i][1][j];
#pragma unroll
            for (int off = 1; off < 16; off <<= 1)   // reduce within 16-lane halves
                s += __shfl_xor(s, off, 32);
            if (lr == 0) {
                int row = wm * 64 + i * 16 + lh * 8 + j;
                sSum[row * 4 + wn] = s;
            }
        }
    }
    __syncthreads();
    if (tid < BM) {
        float t = sSum[tid * 4 + 0] + sSum[tid * 4 + 1] +
                  sSum[tid * 4 + 2] + sSum[tid * 4 + 3];
        sInv[tid] = rsqrtf(t * (1.0f / 128.0f) + 1e-6f);   // group size 128
    }
    __syncthreads();

#pragma unroll
    for (int jn = 0; jn < 2; ++jn) {
        int col = n0 + wn * 32 + jn * 16 + lr;
        float gm = gamma[col];
#pragma unroll
        for (int i = 0; i < 4; ++i) {
            int rloc = wm * 64 + i * 16 + lh * 8;
#pragma unroll
            for (int j = 0; j < 8; ++j)
                out[(size_t)(m0 + rloc + j) * N + col] = acc[i][jn][j] * sInv[rloc + j] * gm;
        }
    }
}

// ===========================================================================
// Fallback kernels (workspace too small for preconverted weights):
// convert f32 -> bf16 inside the GEMM loop (round-2 structure, v_perm pack).
// ===========================================================================

__global__ __launch_bounds__(256)
void dual_gemm_gate(const float* __restrict__ z,
                    const float* __restrict__ W1, const float* __restrict__ b1,
                    const float* __restrict__ W2, const float* __restrict__ b2,
                    unsigned short* __restrict__ gated,
                    int M, int K, int N)
{
    __shared__ __attribute__((aligned(16))) unsigned short sA [2][BM * LP];
    __shared__ __attribute__((aligned(16))) unsigned short sB1[2][BN * LP];
    __shared__ __attribute__((aligned(16))) unsigned short sB2[2][BN * LP];

    const int tid  = threadIdx.x;
    const int lane = tid & 31, wave = tid >> 5;
    const int wm = wave >> 2, wn = wave & 3;
    const int lr = lane & 15, lh = lane >> 4;
    const int m0 = blockIdx.y * BM, n0 = blockIdx.x * BN;

    v8f acc1[4][2], acc2[4][2];
#pragma unroll
    for (int i = 0; i < 4; ++i)
#pragma unroll
        for (int j = 0; j < 2; ++j)
#pragma unroll
            for (int e = 0; e < 8; ++e) { acc1[i][j][e] = 0.f; acc2[i][j][e] = 0.f; }

    float4 aR[4], b1R[4], b2R[4];

    auto gload = [&](int kb) {
#pragma unroll
        for (int r = 0; r < 4; ++r) {
            int idx = tid + r * 256;
            int row = idx >> 3, k4 = idx & 7;
            aR[r] = *reinterpret_cast<const float4*>(
                z + (size_t)(m0 + row) * K + kb + k4 * 4);
        }
#pragma unroll
        for (int p = 0; p < 2; ++p) {
            int idx = tid + p * 256;
            int kr2 = idx >> 5, n4 = idx & 31;
            const size_t g = (size_t)(kb + 2 * kr2) * N + n0 + n4 * 4;
            b1R[2 * p + 0] = *reinterpret_cast<const float4*>(W1 + g);
            b1R[2 * p + 1] = *reinterpret_cast<const float4*>(W1 + g + N);
            b2R[2 * p + 0] = *reinterpret_cast<const float4*>(W2 + g);
            b2R[2 * p + 1] = *reinterpret_cast<const float4*>(W2 + g + N);
        }
    };
    auto lstore = [&](int buf) {
#pragma unroll
        for (int r = 0; r < 4; ++r) {
            int idx = tid + r * 256;
            int row = idx >> 3, k4 = idx & 7;
            uint2 pk;
            pk.x = pack2bf(aR[r].x, aR[r].y);
            pk.y = pack2bf(aR[r].z, aR[r].w);
            *reinterpret_cast<uint2*>(
                reinterpret_cast<char*>(sA[buf]) + row * LPB + k4 * 8) = pk;
        }
#pragma unroll
        for (int p = 0; p < 2; ++p) {
            int idx = tid + p * 256;
            int kr2 = idx >> 5, n4 = idx & 31;
            int nb = n4 * 4, kk = 2 * kr2;
            unsigned* d1 = reinterpret_cast<unsigned*>(&sB1[buf][0]);
            unsigned* d2 = reinterpret_cast<unsigned*>(&sB2[buf][0]);
            d1[((nb + 0) * LP + kk) >> 1] = pack2bf(b1R[2*p].x, b1R[2*p+1].x);
            d1[((nb + 1) * LP + kk) >> 1] = pack2bf(b1R[2*p].y, b1R[2*p+1].y);
            d1[((nb + 2) * LP + kk) >> 1] = pack2bf(b1R[2*p].z, b1R[2*p+1].z);
            d1[((nb + 3) * LP + kk) >> 1] = pack2bf(b1R[2*p].w, b1R[2*p+1].w);
            d2[((nb + 0) * LP + kk) >> 1] = pack2bf(b2R[2*p].x, b2R[2*p+1].x);
            d2[((nb + 1) * LP + kk) >> 1] = pack2bf(b2R[2*p].y, b2R[2*p+1].y);
            d2[((nb + 2) * LP + kk) >> 1] = pack2bf(b2R[2*p].z, b2R[2*p+1].z);
            d2[((nb + 3) * LP + kk) >> 1] = pack2bf(b2R[2*p].w, b2R[2*p+1].w);
        }
    };
    auto compute = [&](int buf) {
        FragB16 aF[4], b1F[2], b2F[2];
#pragma unroll
        for (int i = 0; i < 4; ++i) {
            int row = wm * 64 + i * 16 + lr;
            const char* base = reinterpret_cast<const char*>(sA[buf]) + row * LPB + lh * 32;
            aF[i].u4[0] = *reinterpret_cast<const uint4*>(base);
            aF[i].u4[1] = *reinterpret_cast<const uint4*>(base + 16);
        }
#pragma unroll
        for (int jn = 0; jn < 2; ++jn) {
            int col = wn * 32 + jn * 16 + lr;
            const char* c1 = reinterpret_cast<const char*>(sB1[buf]) + col * LPB + lh * 32;
            b1F[jn].u4[0] = *reinterpret_cast<const uint4*>(c1);
            b1F[jn].u4[1] = *reinterpret_cast<const uint4*>(c1 + 16);
            const char* c2 = reinterpret_cast<const char*>(sB2[buf]) + col * LPB + lh * 32;
            b2F[jn].u4[0] = *reinterpret_cast<const uint4*>(c2);
            b2F[jn].u4[1] = *reinterpret_cast<const uint4*>(c2 + 16);
        }
#pragma unroll
        for (int i = 0; i < 4; ++i)
#pragma unroll
            for (int jn = 0; jn < 2; ++jn) {
                acc1[i][jn] = __builtin_amdgcn_wmma_f32_16x16x32_bf16(
                    false, aF[i].v, false, b1F[jn].v, (short)0, acc1[i][jn], false, false);
                acc2[i][jn] = __builtin_amdgcn_wmma_f32_16x16x32_bf16(
                    false, aF[i].v, false, b2F[jn].v, (short)0, acc2[i][jn], false, false);
            }
    };

    const int nIter = K / BK;
    gload(0);
    lstore(0);
    __syncthreads();
    for (int it = 0; it < nIter; ++it) {
        const int cur = it & 1;
        const bool pre = (it + 1 < nIter);
        if (pre) gload((it + 1) * BK);
        compute(cur);
        if (pre) lstore(cur ^ 1);
        __syncthreads();
    }

#pragma unroll
    for (int jn = 0; jn < 2; ++jn) {
        int col = n0 + wn * 32 + jn * 16 + lr;
        float bb1 = b1[col], bb2 = b2[col];
#pragma unroll
        for (int i = 0; i < 4; ++i) {
            int rbase = m0 + wm * 64 + i * 16 + lh * 8;
#pragma unroll
            for (int j = 0; j < 8; ++j) {
                float g = acc1[i][jn][j] + bb1;
                float c = acc2[i][jn][j] + bb2;
                float val = c * (1.0f / (1.0f + __expf(-g)));
                gated[(size_t)(rbase + j) * N + col] = f2bf(val);
            }
        }
    }
}

__global__ __launch_bounds__(256)
void gemm_rivalry_norm(const unsigned short* __restrict__ gated,
                       const float* __restrict__ W3, const float* __restrict__ b3,
                       const float* __restrict__ gamma,
                       float* __restrict__ out,
                       int M, int K, int N)
{
    __shared__ __attribute__((aligned(16))) unsigned short sA[2][BM * LP];
    __shared__ __attribute__((aligned(16))) unsigned short sB[2][BN * LP];
    __shared__ float sSum[BM * 4];
    __shared__ float sInv[BM];

    const int tid  = threadIdx.x;
    const int lane = tid & 31, wave = tid >> 5;
    const int wm = wave >> 2, wn = wave & 3;
    const int lr = lane & 15, lh = lane >> 4;
    const int m0 = blockIdx.y * BM, n0 = blockIdx.x * BN;

    v8f acc[4][2];
#pragma unroll
    for (int i = 0; i < 4; ++i)
#pragma unroll
        for (int j = 0; j < 2; ++j)
#pragma unroll
            for (int e = 0; e < 8; ++e) acc[i][j][e] = 0.f;

    float4 bR[4];

    auto gloadB = [&](int kb) {
#pragma unroll
        for (int p = 0; p < 2; ++p) {
            int idx = tid + p * 256;
            int kr2 = idx >> 5, n4 = idx & 31;
            const size_t g = (size_t)(kb + 2 * kr2) * N + n0 + n4 * 4;
            bR[2 * p + 0] = *reinterpret_cast<const float4*>(W3 + g);
            bR[2 * p + 1] = *reinterpret_cast<const float4*>(W3 + g + N);
        }
    };
    auto lstoreB = [&](int buf) {
#pragma unroll
        for (int p = 0; p < 2; ++p) {
            int idx = tid + p * 256;
            int kr2 = idx >> 5, n4 = idx & 31;
            int nb = n4 * 4, kk = 2 * kr2;
            unsigned* d = reinterpret_cast<unsigned*>(&sB[buf][0]);
            d[((nb + 0) * LP + kk) >> 1] = pack2bf(bR[2*p].x, bR[2*p+1].x);
            d[((nb + 1) * LP + kk) >> 1] = pack2bf(bR[2*p].y, bR[2*p+1].y);
            d[((nb + 2) * LP + kk) >> 1] = pack2bf(bR[2*p].z, bR[2*p+1].z);
            d[((nb + 3) * LP + kk) >> 1] = pack2bf(bR[2*p].w, bR[2*p+1].w);
        }
    };
    auto compute = [&](int buf) {
        FragB16 aF[4], bF[2];
#pragma unroll
        for (int i = 0; i < 4; ++i) {
            int row = wm * 64 + i * 16 + lr;
            const char* base = reinterpret_cast<const char*>(sA[buf]) + row * LPB + lh * 32;
            aF[i].u4[0] = *reinterpret_cast<const uint4*>(base);
            aF[i].u4[1] = *reinterpret_cast<const uint4*>(base + 16);
        }
#pragma unroll
        for (int jn = 0; jn < 2; ++jn) {
            int col = wn * 32 + jn * 16 + lr;
            const char* cb = reinterpret_cast<const char*>(sB[buf]) + col * LPB + lh * 32;
            bF[jn].u4[0] = *reinterpret_cast<const uint4*>(cb);
            bF[jn].u4[1] = *reinterpret_cast<const uint4*>(cb + 16);
        }
#pragma unroll
        for (int i = 0; i < 4; ++i)
#pragma unroll
            for (int jn = 0; jn < 2; ++jn)
                acc[i][jn] = __builtin_amdgcn_wmma_f32_16x16x32_bf16(
                    false, aF[i].v, false, bF[jn].v, (short)0, acc[i][jn], false, false);
    };

    const int nIter = K / BK;
    {
#pragma unroll
        for (int r = 0; r < 2; ++r) {
            int idx = tid + r * 256;
            int row = idx >> 2, k8 = idx & 3;
            async_copy_b128(reinterpret_cast<char*>(sA[0]) + row * LPB + k8 * 16,
                            gated + (size_t)(m0 + row) * K + k8 * 8);
        }
    }
    gloadB(0);
    lstoreB(0);
    wait_async0();
    __syncthreads();
    for (int it = 0; it < nIter; ++it) {
        const int cur = it & 1;
        const bool pre = (it + 1 < nIter);
        if (pre) {
            const int kbn = (it + 1) * BK;
#pragma unroll
            for (int r = 0; r < 2; ++r) {
                int idx = tid + r * 256;
                int row = idx >> 2, k8 = idx & 3;
                async_copy_b128(reinterpret_cast<char*>(sA[cur ^ 1]) + row * LPB + k8 * 16,
                                gated + (size_t)(m0 + row) * K + kbn + k8 * 8);
            }
            gloadB(kbn);
        }
        compute(cur);
        if (pre) lstoreB(cur ^ 1);
        wait_async0();
        __syncthreads();
    }

#pragma unroll
    for (int jn = 0; jn < 2; ++jn) {
        int col = n0 + wn * 32 + jn * 16 + lr;
        float bb = b3[col];
#pragma unroll
        for (int i = 0; i < 4; ++i)
#pragma unroll
            for (int j = 0; j < 8; ++j) acc[i][jn][j] += bb;
    }
#pragma unroll
    for (int i = 0; i < 4; ++i) {
#pragma unroll
        for (int j = 0; j < 8; ++j) {
            float s = acc[i][0][j] * acc[i][0][j] + acc[i][1][j] * acc[i][1][j];
#pragma unroll
            for (int off = 1; off < 16; off <<= 1)
                s += __shfl_xor(s, off, 32);
            if (lr == 0) {
                int row = wm * 64 + i * 16 + lh * 8 + j;
                sSum[row * 4 + wn] = s;
            }
        }
    }
    __syncthreads();
    if (tid < BM) {
        float t = sSum[tid * 4 + 0] + sSum[tid * 4 + 1] +
                  sSum[tid * 4 + 2] + sSum[tid * 4 + 3];
        sInv[tid] = rsqrtf(t * (1.0f / 128.0f) + 1e-6f);
    }
    __syncthreads();
#pragma unroll
    for (int jn = 0; jn < 2; ++jn) {
        int col = n0 + wn * 32 + jn * 16 + lr;
        float gm = gamma[col];
#pragma unroll
        for (int i = 0; i < 4; ++i) {
            int rloc = wm * 64 + i * 16 + lh * 8;
#pragma unroll
            for (int j = 0; j < 8; ++j)
                out[(size_t)(m0 + rloc + j) * N + col] = acc[i][jn][j] * sInv[rloc + j] * gm;
        }
    }
}

// ---------------------------------------------------------------------------
extern "C" void kernel_launch(void* const* d_in, const int* in_sizes, int n_in,
                              void* d_out, int out_size, void* d_ws, size_t ws_size,
                              hipStream_t stream)
{
    const float* z     = (const float*)d_in[0];
    const float* W1    = (const float*)d_in[1];
    const float* b1    = (const float*)d_in[2];
    const float* W2    = (const float*)d_in[3];
    const float* b2    = (const float*)d_in[4];
    const float* W3    = (const float*)d_in[5];
    const float* b3    = (const float*)d_in[6];
    const float* gamma = (const float*)d_in[7];

    const int Dmid = in_sizes[2];                 // 8192
    const int Dout = in_sizes[6];                 // 4096
    const int Din  = in_sizes[1] / Dmid;          // 4096
    const int Mtok = in_sizes[0] / Din;           // 8192 (= B*S)

    const size_t szZb  = (size_t)Mtok * Din;      // bf16 elem counts
    const size_t szW12 = (size_t)Din * Dmid;
    const size_t szW3  = (size_t)Dmid * Dout;
    const size_t szGat = (size_t)Mtok * Dmid;
    const size_t needFast = (szZb + 2 * szW12 + szW3 + szGat) * sizeof(unsigned short);

    dim3 blk(256);

    if (ws_size >= needFast) {
        // -------- fast path: preconvert everything to bf16, async-DMA GEMMs --------
        unsigned short* zb  = (unsigned short*)d_ws;
        unsigned short* w1t = zb  + szZb;
        unsigned short* w2t = w1t + szW12;
        unsigned short* w3t = w2t + szW12;
        unsigned short* gat = w3t + szW3;

        convert_bf16<<<(unsigned)(szZb / 1024), blk, 0, stream>>>(z, zb);
        transpose_to_bf16<<<dim3(Dmid / 64, Din / 64), blk, 0, stream>>>(W1, w1t, Din, Dmid);
        transpose_to_bf16<<<dim3(Dmid / 64, Din / 64), blk, 0, stream>>>(W2, w2t, Din, Dmid);
        transpose_to_bf16<<<dim3(Dout / 64, Dmid / 64), blk, 0, stream>>>(W3, w3t, Dmid, Dout);

        dual_gemm_gate_fast<<<dim3(Dmid / BN, Mtok / BM), blk, 0, stream>>>(
            zb, w1t, w2t, b1, b2, gat, Mtok, Din, Dmid);
        gemm_norm_fast<<<dim3(Dout / BN, Mtok / BM), blk, 0, stream>>>(
            gat, w3t, b3, gamma, (float*)d_out, Mtok, Dmid, Dout);
    } else {
        // -------- fallback: convert inside the GEMM loop --------
        unsigned short* gat = (unsigned short*)d_ws;
        dual_gemm_gate<<<dim3(Dmid / BN, Mtok / BM), blk, 0, stream>>>(
            z, W1, b1, W2, b2, gat, Mtok, Din, Dmid);
        gemm_rivalry_norm<<<dim3(Dout / BN, Mtok / BM), blk, 0, stream>>>(
            gat, W3, b3, gamma, (float*)d_out, Mtok, Dmid, Dout);
    }
}
